// AxialAttention_dynamic_Block_37769942401251
// MI455X (gfx1250) — matrix operand order repaired and checked by reading, not compiled, not run on hardware
//
#include <hip/hip_runtime.h>
#include <math.h>

typedef __attribute__((ext_vector_type(16))) __bf16 v16bf;
typedef __attribute__((ext_vector_type(8)))  float  v8f;
typedef __attribute__((ext_vector_type(8)))  unsigned short us8;

#define DEV static __device__ __forceinline__

DEV unsigned short f2bf(float f){
  unsigned int u = __builtin_bit_cast(unsigned int, f);
  u = (u + 0x7FFFu + ((u >> 16) & 1u)) >> 16;
  return (unsigned short)u;
}
DEV float bf2f(unsigned short s){
  unsigned int u = ((unsigned int)s) << 16;
  return __builtin_bit_cast(float, u);
}
DEV __bf16 usb(unsigned short s){ return __builtin_bit_cast(__bf16, s); }

// A fragment (16x32 bf16). rowbase = buf + row*ld + kc for this lane's row (row = tile_row + (lane&15)).
// ISA layout: lane L (row = L&15) holds k = ((h>>3)<<4) + ((L>>4)<<3) + (h&7)  -> two contiguous 8-chunks.
DEV v16bf load_a(const unsigned short* rowbase, int lane){
  const us8* p0 = (const us8*)(rowbase + ((lane >> 4) << 3));
  const us8* p1 = (const us8*)(rowbase + 16 + ((lane >> 4) << 3));
  us8 u0 = *p0, u1 = *p1;
  v16bf a;
#pragma unroll
  for (int h = 0; h < 8; ++h){ a[h] = usb(u0[h]); a[h + 8] = usb(u1[h]); }
  return a;
}
// B fragment (32x16 bf16). colbase = buf + n*ld + kc (n = tile_col + (lane&15), k contiguous in source).
// ISA layout: lane L (col = L&15) holds k = ((h>>3)<<4) + ((L>>4)<<3) + (h&7)  -> two contiguous 8-chunks.
DEV v16bf load_b(const unsigned short* colbase, int lane){
  const us8* p0 = (const us8*)(colbase + ((lane >> 4) << 3));
  const us8* p1 = (const us8*)(colbase + 16 + ((lane >> 4) << 3));
  us8 u0 = *p0, u1 = *p1;
  v16bf b;
#pragma unroll
  for (int h = 0; h < 8; ++h){ b[h] = usb(u0[h]); b[h + 8] = usb(u1[h]); }
  return b;
}
DEV v8f wmma_bf16(v16bf a, v16bf b, v8f c){
  return __builtin_amdgcn_wmma_f32_16x16x32_bf16(false, a, false, b, (short)0, c, false, false);
}

// ---------------- conversion kernels ----------------
__global__ void k_cvt(const float* __restrict__ src, unsigned short* __restrict__ dst){
  int i = blockIdx.x * 256 + threadIdx.x;
  dst[i] = f2bf(src[i]);
}
// relT[t][c] (512 x 128, c contiguous): B-operand source for Pq/Pk (k = c contiguous per column t)
__global__ void k_cvt_relT(const float* __restrict__ rel, unsigned short* __restrict__ relT){
  int i = blockIdx.x * 256 + threadIdx.x; // 65536
  int t = i >> 7, c = i & 127;
  relT[i] = (t < 511) ? f2bf(rel[(size_t)c * 511 + t]) : (unsigned short)0;
}
// relV[c][t] (64 x 512, t contiguous): B-operand source for sve (k = t contiguous per column c)
__global__ void k_cvt_relV(const float* __restrict__ rel, unsigned short* __restrict__ relV){
  int i = blockIdx.x * 256 + threadIdx.x; // 32768
  int c = i >> 9, t = i & 511;
  relV[i] = (t < 511) ? f2bf(rel[(size_t)(64 + c) * 511 + t]) : (unsigned short)0;
}

// ---------------- qkv GEMM: qkv[b,o,n] = sum_c w[o,c]*x[b,n,c] ----------------
// Ping-pong software pipeline (unroll-by-2): two named fragment sets, each half of the
// loop loads the other set one K-step ahead -> no rotation moves, loads overlap WMMAs.
__global__ void k_qkv(const unsigned short* __restrict__ xbf,
                      const unsigned short* __restrict__ wbf,
                      float* __restrict__ qkv){
  int bid = blockIdx.x;           // 2048 = 64 b * 16 o-tiles(64) * 2 n-tiles(128)
  int b  = bid >> 5;
  int r5 = bid & 31;
  int o0 = (r5 >> 1) << 6;
  int n0 = (r5 & 1) << 7;
  int tid = threadIdx.x, lane = tid & 31, w = tid >> 5;
  int ow = o0 + ((w >> 1) << 4);  // 16 o-rows per wave
  int nw = n0 + ((w & 1) << 6);   // 64 n-cols per wave
  v8f acc[4] = {{}, {}, {}, {}};
  const unsigned short* xb   = xbf + (size_t)b * 256 * 512;
  const unsigned short* arow = &wbf[(size_t)(ow + (lane & 15)) * 512];
  const unsigned short* brow[4];
#pragma unroll
  for (int t = 0; t < 4; ++t) brow[t] = &xb[(size_t)(nw + t * 16 + (lane & 15)) * 512];

  v16bf a0 = load_a(arow, lane), a1;
  v16bf b0[4], b1[4];
#pragma unroll
  for (int t = 0; t < 4; ++t) b0[t] = load_b(brow[t], lane);
#pragma unroll 1
  for (int kc = 0; kc < 448; kc += 64){
    a1 = load_a(arow + kc + 32, lane);
#pragma unroll
    for (int t = 0; t < 4; ++t) b1[t] = load_b(brow[t] + kc + 32, lane);
#pragma unroll
    for (int t = 0; t < 4; ++t) acc[t] = wmma_bf16(a0, b0[t], acc[t]);
    a0 = load_a(arow + kc + 64, lane);
#pragma unroll
    for (int t = 0; t < 4; ++t) b0[t] = load_b(brow[t] + kc + 64, lane);
#pragma unroll
    for (int t = 0; t < 4; ++t) acc[t] = wmma_bf16(a1, b1[t], acc[t]);
  }
  a1 = load_a(arow + 480, lane);
#pragma unroll
  for (int t = 0; t < 4; ++t) b1[t] = load_b(brow[t] + 480, lane);
#pragma unroll
  for (int t = 0; t < 4; ++t) acc[t] = wmma_bf16(a0, b0[t], acc[t]);
#pragma unroll
  for (int t = 0; t < 4; ++t) acc[t] = wmma_bf16(a1, b1[t], acc[t]);

#pragma unroll
  for (int t = 0; t < 4; ++t)
#pragma unroll
    for (int r = 0; r < 8; ++r){
      int o = ow + r + ((lane >> 4) << 3);
      int n = nw + t * 16 + (lane & 15);
      qkv[((size_t)(b * 1024 + o)) * 256 + n] = acc[t][r];
    }
}

// ---------------- per-channel BN stats (1024 channels over B*N) ----------------
__global__ void k_chan_stats(const float* __restrict__ buf, const float* __restrict__ gv,
                             const float* __restrict__ bv,
                             float* __restrict__ scale, float* __restrict__ shift){
  __shared__ float rs[256], rq[256];
  int ch = blockIdx.x;
  int tid = threadIdx.x;
  float s = 0.f, ss = 0.f;
  const float* p = buf + (size_t)ch * 256;
  for (int e = tid; e < 64 * 256; e += 256){
    int bb = e >> 8, n = e & 255;
    float v = p[(size_t)bb * 1024 * 256 + n];
    s += v; ss += v * v;
  }
  rs[tid] = s; rq[tid] = ss; __syncthreads();
  for (int off = 128; off; off >>= 1){
    if (tid < off){ rs[tid] += rs[tid + off]; rq[tid] += rq[tid + off]; }
    __syncthreads();
  }
  if (tid == 0){
    float cnt  = 16384.0f;
    float mean = rs[0] / cnt;
    float var  = rq[0] / cnt - mean * mean;
    float sc   = gv[ch] * rsqrtf(var + 1e-5f);
    scale[ch] = sc;
    shift[ch] = bv[ch] - mean * sc;
  }
}

// ---------------- k_sim: qk, Pq, Pk per (b,g,64-row strip) + masked stat partials ----------------
__global__ void k_sim(const float* __restrict__ qkv,
                      const float* __restrict__ qscale, const float* __restrict__ qshift,
                      const unsigned short* __restrict__ relT,
                      unsigned short* __restrict__ qk_s,
                      unsigned short* __restrict__ Pq_s,
                      unsigned short* __restrict__ Pk_s,
                      float* __restrict__ part){
  __shared__ unsigned short qT[64 * 32];   // q strip, [i][c]
  __shared__ unsigned short kT[256 * 32];  // full k,  [j][c]
  __shared__ float wred[8][6];
  int bid = blockIdx.x;           // 2048 = 64 b * 8 g * 4 strips
  int b  = bid >> 5;
  int g  = (bid >> 2) & 7;
  int is = bid & 3;
  int i0 = is << 6;
  int tid = threadIdx.x, lane = tid & 31, w = tid >> 5;
  int bg = b * 8 + g;
  for (int e = tid; e < 64 * 32; e += 256){
    int c = e >> 6, i = e & 63;
    int ch = g * 128 + c;
    float v = qkv[(size_t)(b * 1024 + ch) * 256 + i0 + i] * qscale[ch] + qshift[ch];
    qT[i * 32 + c] = f2bf(v);
  }
  for (int e = tid; e < 256 * 32; e += 256){
    int c = e >> 8, j = e & 255;
    int ch = g * 128 + 32 + c;
    float v = qkv[(size_t)(b * 1024 + ch) * 256 + j] * qscale[ch] + qshift[ch];
    kT[j * 32 + c] = f2bf(v);
  }
  __syncthreads();
  float s_qk = 0, s2_qk = 0, s_qr = 0, s2_qr = 0, s_kr = 0, s2_kr = 0;

  // qk: 4 i-tiles x 16 j-tiles, K=32 single WMMA; ping-pong prefetch of A/B fragments
  {
#define QK_A(s)  load_a(&qT[((((w + ((s) << 3)) >> 4)) * 16 + (lane & 15)) * 32], lane)
#define QK_B(s)  load_b(&kT[(((w + ((s) << 3)) & 15) * 16 + (lane & 15)) * 32], lane)
#define QK_EMIT(s, av, bv_)                                              \
    {                                                                    \
      int q = w + ((s) << 3);                                            \
      int it = q >> 4, jt = q & 15;                                      \
      v8f acc = {};                                                      \
      acc = wmma_bf16(av, bv_, acc);                                     \
      _Pragma("unroll")                                                  \
      for (int r = 0; r < 8; ++r){                                       \
        int m = r + ((lane >> 4) << 3);                                  \
        int i = i0 + it * 16 + m;                                        \
        int j = jt * 16 + (lane & 15);                                   \
        float v = acc[r];                                                \
        qk_s[((size_t)bg * 256 + i) * 256 + j] = f2bf(v);                \
        s_qk += v; s2_qk += v * v;                                       \
      }                                                                  \
    }
    v16bf a0 = QK_A(0), b0 = QK_B(0), a1, b1;
#pragma unroll 1
    for (int s = 0; s < 6; s += 2){
      a1 = QK_A(s + 1); b1 = QK_B(s + 1);
      QK_EMIT(s, a0, b0);
      a0 = QK_A(s + 2); b0 = QK_B(s + 2);
      QK_EMIT(s + 1, a1, b1);
    }
    a1 = QK_A(7); b1 = QK_B(7);
    QK_EMIT(6, a0, b0);
    QK_EMIT(7, a1, b1);
  }
  // Pq = q^T x rel_q : 4 i-tiles x 32 t-tiles; ping-pong prefetch
  {
#define PQ_A(s)  load_a(&qT[((((w * 16 + (s)) & 3)) * 16 + (lane & 15)) * 32], lane)
#define PQ_B(s)  load_b(&relT[(size_t)(((w * 16 + (s)) >> 2) * 16 + (lane & 15)) * 128 + 0], lane)
#define PQ_EMIT(s, av, bv_)                                              \
    {                                                                    \
      int q = w * 16 + (s);                                              \
      int it = q & 3, tt = q >> 2;                                       \
      v8f acc = {};                                                      \
      acc = wmma_bf16(av, bv_, acc);                                     \
      _Pragma("unroll")                                                  \
      for (int r = 0; r < 8; ++r){                                       \
        int m = r + ((lane >> 4) << 3);                                  \
        int i = i0 + it * 16 + m;                                        \
        int t = tt * 16 + (lane & 15);                                   \
        float v = acc[r];                                                \
        Pq_s[((size_t)bg * 256 + i) * 512 + t] = f2bf(v);                \
        int j = i + 255 - t;                                             \
        if (j >= 0 && j < 256){ float sv = 0.1f * v; s_qr += sv; s2_qr += sv * sv; } \
      }                                                                  \
    }
    v16bf a0 = PQ_A(0), b0 = PQ_B(0), a1, b1;
#pragma unroll 1
    for (int s = 0; s < 14; s += 2){
      a1 = PQ_A(s + 1); b1 = PQ_B(s + 1);
      PQ_EMIT(s, a0, b0);
      a0 = PQ_A(s + 2); b0 = PQ_B(s + 2);
      PQ_EMIT(s + 1, a1, b1);
    }
    a1 = PQ_A(15); b1 = PQ_B(15);
    PQ_EMIT(14, a0, b0);
    PQ_EMIT(15, a1, b1);
  }
  // Pk = k^T x rel_k : rows j in strip; ping-pong prefetch
  {
#define PK_A(s)  load_a(&kT[(size_t)(i0 + (((w * 16 + (s)) & 3)) * 16 + (lane & 15)) * 32], lane)
#define PK_B(s)  load_b(&relT[(size_t)(((w * 16 + (s)) >> 2) * 16 + (lane & 15)) * 128 + 32], lane)
#define PK_EMIT(s, av, bv_)                                              \
    {                                                                    \
      int q = w * 16 + (s);                                              \
      int it = q & 3, tt = q >> 2;                                       \
      v8f acc = {};                                                      \
      acc = wmma_bf16(av, bv_, acc);                                     \
      _Pragma("unroll")                                                  \
      for (int r = 0; r < 8; ++r){                                       \
        int m = r + ((lane >> 4) << 3);                                  \
        int jrow = i0 + it * 16 + m;                                     \
        int t = tt * 16 + (lane & 15);                                   \
        float v = acc[r];                                                \
        Pk_s[((size_t)bg * 256 + jrow) * 512 + t] = f2bf(v);             \
        int i = jrow + 255 - t;                                          \
        if (i >= 0 && i < 256){ float sv = 0.1f * v; s_kr += sv; s2_kr += sv * sv; } \
      }                                                                  \
    }
    v16bf a0 = PK_A(0), b0 = PK_B(0), a1, b1;
#pragma unroll 1
    for (int s = 0; s < 14; s += 2){
      a1 = PK_A(s + 1); b1 = PK_B(s + 1);
      PK_EMIT(s, a0, b0);
      a0 = PK_A(s + 2); b0 = PK_B(s + 2);
      PK_EMIT(s + 1, a1, b1);
    }
    a1 = PK_A(15); b1 = PK_B(15);
    PK_EMIT(14, a0, b0);
    PK_EMIT(15, a1, b1);
  }
  float st[6] = {s_qk, s2_qk, s_qr, s2_qr, s_kr, s2_kr};
#pragma unroll
  for (int k = 0; k < 6; ++k){
    float v = st[k];
    for (int off = 16; off; off >>= 1) v += __shfl_xor(v, off, 32);
    if (lane == 0) wred[w][k] = v;
  }
  __syncthreads();
  if (tid < 6){
    float acc = 0;
    for (int ww = 0; ww < 8; ++ww) acc += wred[ww][tid];
    part[(size_t)bid * 6 + tid] = acc;
  }
}

// ---------------- finalize 24 sim rstd scales (means cancel in softmax) ----------------
__global__ void k_sim_fin(const float* __restrict__ part, const float* __restrict__ g_sim,
                          float* __restrict__ scale24){
  int t = threadIdx.x;
  if (t >= 24) return;
  int comp = t >> 3, g = t & 7;
  float s = 0, ss = 0;
  for (int b = 0; b < 64; ++b)
    for (int is = 0; is < 4; ++is){
      int bid = ((b * 8 + g) << 2) + is;
      s  += part[(size_t)bid * 6 + comp * 2 + 0];
      ss += part[(size_t)bid * 6 + comp * 2 + 1];
    }
  float cnt  = 4194304.0f; // B*N*N
  float mean = s / cnt;
  float var  = ss / cnt - mean * mean;
  scale24[t] = g_sim[t] * rsqrtf(var + 1e-5f);
}

// ---------------- k_attn: sim assembly + softmax + sv/sve WMMAs ----------------
__global__ void k_attn(const float* __restrict__ qkv,
                       const float* __restrict__ qscale, const float* __restrict__ qshift,
                       const unsigned short* __restrict__ qk_s,
                       const unsigned short* __restrict__ Pq_s,
                       const unsigned short* __restrict__ Pk_s,
                       const unsigned short* __restrict__ relV,
                       const float* __restrict__ scale24,
                       float* __restrict__ outpre){
  __shared__ unsigned short sim_l[32 * 256]; // [i_local][j]
  __shared__ unsigned short v_l[64 * 256];   // [c][j]
  int bid = blockIdx.x;          // 4096 = 64 b * 8 g * 8 strips(32)
  int b  = bid >> 6;
  int g  = (bid >> 3) & 7;
  int is = bid & 7;
  int i0 = is << 5;
  int tid = threadIdx.x, lane = tid & 31, w = tid >> 5;
  int bg = b * 8 + g;
  for (int e = tid; e < 64 * 256; e += 256){
    int c = e >> 8, j = e & 255;
    int ch = g * 128 + 64 + c;
    float v = qkv[(size_t)(b * 1024 + ch) * 256 + j] * qscale[ch] + qshift[ch];
    v_l[c * 256 + j] = f2bf(v);
  }
  float aqk = scale24[g];
  float aqr = scale24[8 + g] * 0.1f;
  float akr = scale24[16 + g] * 0.1f;
  // wave w handles rows 4w..4w+3 of the 32-row strip
  for (int rl = 0; rl < 4; ++rl){
    int il = (w << 2) + rl;
    int i  = i0 + il;
    float vals[8];
    float mx = -3.0e38f;
    const unsigned short* qkrow = qk_s + ((size_t)bg * 256 + i) * 256;
    const unsigned short* pqrow = Pq_s + ((size_t)bg * 256 + i) * 512;
#pragma unroll
    for (int cc = 0; cc < 8; ++cc){
      int j = (cc << 5) + lane;
      float v = aqk * bf2f(qkrow[j])
              + aqr * bf2f(pqrow[i + 255 - j])
              + akr * bf2f(Pk_s[((size_t)bg * 256 + j) * 512 + (j + 255 - i)]);
      vals[cc] = v;
      mx = fmaxf(mx, v);
    }
    for (int off = 16; off; off >>= 1) mx = fmaxf(mx, __shfl_xor(mx, off, 32));
    float sum = 0;
#pragma unroll
    for (int cc = 0; cc < 8; ++cc){ vals[cc] = __expf(vals[cc] - mx); sum += vals[cc]; }
    for (int off = 16; off; off >>= 1) sum += __shfl_xor(sum, off, 32);
    float inv = 1.0f / sum;
#pragma unroll
    for (int cc = 0; cc < 8; ++cc){
      int j = (cc << 5) + lane;
      sim_l[il * 256 + j] = f2bf(vals[cc] * inv);
    }
  }
  __syncthreads();
  int it = w >> 2, ct = w & 3;  // 2 i-tiles x 4 c-tiles = 8 waves
  // sv = sim x v^T  (K = j = 256); ping-pong double buffer, no rotation moves
  {
    const unsigned short* arow = &sim_l[(it * 16 + (lane & 15)) * 256];
    const unsigned short* brow = &v_l[(ct * 16 + (lane & 15)) * 256];
    v8f acc = {};
    v16bf a0 = load_a(arow, lane), b0 = load_b(brow, lane), a1, b1;
#pragma unroll 1
    for (int jc = 0; jc < 192; jc += 64){
      a1 = load_a(arow + jc + 32, lane); b1 = load_b(brow + jc + 32, lane);
      acc = wmma_bf16(a0, b0, acc);
      a0 = load_a(arow + jc + 64, lane); b0 = load_b(brow + jc + 64, lane);
      acc = wmma_bf16(a1, b1, acc);
    }
    a1 = load_a(arow + 224, lane); b1 = load_b(brow + 224, lane);
    acc = wmma_bf16(a0, b0, acc);
    acc = wmma_bf16(a1, b1, acc);
#pragma unroll
    for (int r = 0; r < 8; ++r){
      int m = r + ((lane >> 4) << 3);
      int i = i0 + it * 16 + m;
      int c = ct * 16 + (lane & 15);
      int oc = g * 128 + 2 * c;
      outpre[((size_t)(b * 1024 + oc)) * 256 + i] = acc[r]; // F_SV = 1.0
    }
  }
  // sve = S' x rel_v^T  (K = t = 512), S'[i,t] = sim[i, i+255-t] gathered from LDS;
  // global relV B fragments ping-ponged one K-step ahead so they overlap the gathers.
  {
    const unsigned short* brow = &relV[(size_t)(ct * 16 + (lane & 15)) * 512];
    int ilocal = it * 16 + (lane & 15);
    int irow   = i0 + ilocal;
    v8f acc = {};
#define SVE_GATHER(av, tcv)                                               \
    {                                                                     \
      _Pragma("unroll")                                                   \
      for (int h = 0; h < 16; ++h){                                       \
        int t = (tcv) + ((h >> 3) << 4) + ((lane >> 4) << 3) + (h & 7);   \
        int j = irow + 255 - t;                                           \
        unsigned short u = (j >= 0 && j < 256) ? sim_l[ilocal * 256 + j] : (unsigned short)0; \
        av[h] = usb(u);                                                   \
      }                                                                   \
    }
    v16bf b0 = load_b(brow, lane), b1, a;
#pragma unroll 1
    for (int tc = 0; tc < 448; tc += 64){
      b1 = load_b(brow + tc + 32, lane);
      SVE_GATHER(a, tc);
      acc = wmma_bf16(a, b0, acc);
      b0 = load_b(brow + tc + 64, lane);
      SVE_GATHER(a, tc + 32);
      acc = wmma_bf16(a, b1, acc);
    }
    b1 = load_b(brow + 480, lane);
    SVE_GATHER(a, 448);
    acc = wmma_bf16(a, b0, acc);
    SVE_GATHER(a, 480);
    acc = wmma_bf16(a, b1, acc);
#pragma unroll
    for (int r = 0; r < 8; ++r){
      int m = r + ((lane >> 4) << 3);
      int i = i0 + it * 16 + m;
      int c = ct * 16 + (lane & 15);
      int oc = g * 128 + 2 * c + 1;
      outpre[((size_t)(b * 1024 + oc)) * 256 + i] = 0.1f * acc[r]; // F_SVE
    }
  }
}

// ---------------- final: out[b,o,n] = BN(pre[2o]) + BN(pre[2o+1]) ----------------
__global__ void k_out_final(const float* __restrict__ pre, const float* __restrict__ sc,
                            const float* __restrict__ sh, float* __restrict__ out){
  int idx = blockIdx.x * 256 + threadIdx.x; // 8388608
  int n = idx & 255;
  int o = (idx >> 8) & 511;
  int b = idx >> 17;
  size_t base = ((size_t)(b * 1024 + 2 * o)) * 256 + n;
  float v0 = pre[base]       * sc[2 * o]     + sh[2 * o];
  float v1 = pre[base + 256] * sc[2 * o + 1] + sh[2 * o + 1];
  out[(size_t)idx] = v0 + v1;
}

extern "C" void kernel_launch(void* const* d_in, const int* in_sizes, int n_in,
                              void* d_out, int out_size, void* d_ws, size_t ws_size,
                              hipStream_t stream){
  (void)in_sizes; (void)n_in; (void)out_size; (void)ws_size;
  const float* x     = (const float*)d_in[0];
  const float* w_qkv = (const float*)d_in[1];
  const float* rel   = (const float*)d_in[2];
  const float* g_qkv = (const float*)d_in[3];
  const float* b_qkv = (const float*)d_in[4];
  const float* g_sim = (const float*)d_in[5];
  const float* b_sim = (const float*)d_in[6]; (void)b_sim; // cancels in softmax
  const float* g_out = (const float*)d_in[7];
  const float* b_out = (const float*)d_in[8];
  float* out = (float*)d_out;
  char* ws = (char*)d_ws;

  unsigned short* xbf  = (unsigned short*)(ws + 0);           // 16,777,216 B
  unsigned short* wbf  = (unsigned short*)(ws + 16777216);    //  1,048,576 B
  unsigned short* relT = (unsigned short*)(ws + 17825792);    //    131,072 B
  unsigned short* relV = (unsigned short*)(ws + 17956864);    //     65,536 B
  float* qkv           = (float*)(ws + 18022400);             // 67,108,864 B
  float* qscale        = (float*)(ws + 85131264);             //      8,192 B
  float* qshift        = qscale + 1024;
  unsigned short* qk_s = (unsigned short*)(ws + 85139456);    // 67,108,864 B
  unsigned short* Pq_s = (unsigned short*)(ws + 152248320);   // 134,217,728 B
  unsigned short* Pk_s = (unsigned short*)(ws + 286466048);   // 134,217,728 B
  float* part          = (float*)(ws + 420683776);            //     49,152 B
  float* scale24       = (float*)(ws + 420732928);            //      1,024 B
  float* outpre        = (float*)(ws + 420733952);            // 67,108,864 B
  float* oscale        = (float*)(ws + 487842816);            //      8,192 B
  float* oshift        = oscale + 1024;

  k_cvt<<<32768, 256, 0, stream>>>(x, xbf);
  k_cvt<<<2048, 256, 0, stream>>>(w_qkv, wbf);
  k_cvt_relT<<<256, 256, 0, stream>>>(rel, relT);
  k_cvt_relV<<<128, 256, 0, stream>>>(rel, relV);
  k_qkv<<<2048, 256, 0, stream>>>(xbf, wbf, qkv);
  k_chan_stats<<<1024, 256, 0, stream>>>(qkv, g_qkv, b_qkv, qscale, qshift);
  k_sim<<<2048, 256, 0, stream>>>(qkv, qscale, qshift, relT, qk_s, Pq_s, Pk_s, part);
  k_sim_fin<<<1, 32, 0, stream>>>(part, g_sim, scale24);
  k_attn<<<4096, 256, 0, stream>>>(qkv, qscale, qshift, qk_s, Pq_s, Pk_s, relV, scale24, outpre);
  k_chan_stats<<<1024, 256, 0, stream>>>(outpre, g_out, b_out, oscale, oshift);
  k_out_final<<<32768, 256, 0, stream>>>(outpre, oscale, oshift, out);
}
